// ElmoEmbeddings_78907139162806
// MI455X (gfx1250) — matrix operand order repaired and checked by reading, not compiled, not run on hardware
//
#include <hip/hip_runtime.h>
#include <hip/hip_bf16.h>

// ---------------------------------------------------------------------------
// ELMo LSTM-with-projection for MI455X (gfx1250), wave32 + WMMA bf16.
//   B=32, S=128, D=512, UNITS=4096, PROJ=512
// Strategy:
//   phase0: convert/transpose x -> bf16 [S*B, D]; pack W,R,P into per-lane
//           WMMA B-fragment order (bf16); zero c-state and h.
//   phase1: Zpre[t,b,4U] = x@W + bias  (one big WMMA GEMM, stored bf16)
//   loop t: gates kernel  (h@R via WMMA, cell update, unclipped-c tanh,
//                          cell clip, h_full bf16)
//           proj  kernel  (h_full@P via WMMA, K split over 8 waves + LDS
//                          reduce, clip, write d_out + next-step h bf16)
// ---------------------------------------------------------------------------

#define B_     32
#define S_     128
#define D_     512
#define UNITS_ 4096
#define PROJ_  512
#define G4_    (4 * UNITS_)   // 16384

typedef __attribute__((ext_vector_type(16))) __bf16 bf16x16;
typedef __attribute__((ext_vector_type(8)))  __bf16 bf16x8;
typedef __attribute__((ext_vector_type(8)))  float  f32x8;

// Hardware TANH transcendental if the toolchain exposes it; OCML otherwise.
#if __has_builtin(__builtin_amdgcn_tanhf)
__device__ __forceinline__ float fast_tanh(float x) { return __builtin_amdgcn_tanhf(x); }
#elif __has_builtin(__builtin_amdgcn_tanh_f32)
__device__ __forceinline__ float fast_tanh(float x) { return __builtin_amdgcn_tanh_f32(x); }
#else
__device__ __forceinline__ float fast_tanh(float x) { return tanhf(x); }
#endif

__device__ __forceinline__ bf16x16 cat8(bf16x8 lo, bf16x8 hi) {
    return __builtin_shufflevector(lo, hi, 0,1,2,3,4,5,6,7,8,9,10,11,12,13,14,15);
}

// A-matrix 16x32 bf16 fragment (ISA 7.12.2 layout):
//  lane l (row = row0 + l%16): elems 0..7  = K[k0 + (l>>4)*8 ..]
//                              elems 8..15 = K[k0 + 16 + (l>>4)*8 ..]
__device__ __forceinline__ bf16x16 load_a_frag(const __bf16* __restrict__ base,
                                               int rowStride, int row0, int k0, int lane) {
    int r   = row0 + (lane & 15);
    int sel = lane >> 4;
    const __bf16* p = base + (size_t)r * rowStride + k0 + sel * 8;
    bf16x8 lo = *(const bf16x8*)(p);
    bf16x8 hi = *(const bf16x8*)(p + 16);
    return cat8(lo, hi);
}

// Packed B fragment: 512 bf16 (1 KB), lane l owns bytes [l*32, l*32+32)
__device__ __forceinline__ bf16x16 load_b_frag(const __bf16* __restrict__ frag, int lane) {
    const __bf16* p = frag + lane * 16;
    return cat8(*(const bf16x8*)(p), *(const bf16x8*)(p + 8));
}

__device__ __forceinline__ f32x8 wmma_bf16(bf16x16 a, bf16x16 b, f32x8 c) {
    return __builtin_amdgcn_wmma_f32_16x16x32_bf16(false, a, false, b, (short)0, c, false, false);
}

__device__ __forceinline__ float hsig(float x) {
    return __builtin_fminf(__builtin_fmaxf(__builtin_fmaf(0.2f, x, 0.5f), 0.0f), 1.0f);
}
__device__ __forceinline__ float clip3(float x) {
    return __builtin_fminf(__builtin_fmaxf(x, -3.0f), 3.0f);
}

// ----------------------------- phase 0 -------------------------------------

// x [B,S,D] f32 -> xbf [S*B, D] bf16  (row = t*B + b)
__global__ void xprep_kernel(const float* __restrict__ x, __bf16* __restrict__ xbf) {
    size_t total = (size_t)S_ * B_ * D_;
    for (size_t e = (size_t)blockIdx.x * blockDim.x + threadIdx.x; e < total;
         e += (size_t)gridDim.x * blockDim.x) {
        int d = (int)(e % D_);
        size_t r = e / D_;
        int t = (int)(r / B_);
        int b = (int)(r % B_);
        xbf[e] = (__bf16)x[((size_t)b * S_ + t) * D_ + d];
    }
}

// Pack row-major W[K][N] f32 into bf16 WMMA-B fragment order.
// Fragment f = nt*KT + kt; element e in frag: lane = (e%512)/16, j = e%16;
// k = kt*32 + (j<8?0:16) + (lane>>4)*8 + (j&7); n = nt*16 + lane%16.
__global__ void pack_b_kernel(const float* __restrict__ W, __bf16* __restrict__ pk,
                              int K, int N) {
    int KT = K >> 5;
    size_t total = (size_t)K * N;
    for (size_t e = (size_t)blockIdx.x * blockDim.x + threadIdx.x; e < total;
         e += (size_t)gridDim.x * blockDim.x) {
        size_t frag = e >> 9;
        int rem  = (int)(e & 511);
        int lane = rem >> 4;
        int j    = rem & 15;
        int kt = (int)(frag % KT);
        int nt = (int)(frag / KT);
        int kl = ((j < 8) ? 0 : 16) + ((lane >> 4) << 3) + (j & 7);
        int k  = (kt << 5) + kl;
        int n  = (nt << 4) + (lane & 15);
        pk[e] = (__bf16)W[(size_t)k * N + n];
    }
}

__global__ void init_state_kernel(float* __restrict__ cst, __bf16* __restrict__ hbf) {
    int tid = blockIdx.x * blockDim.x + threadIdx.x;
    int stride = gridDim.x * blockDim.x;
    for (int i = tid; i < B_ * UNITS_; i += stride) cst[i] = 0.0f;
    for (int i = tid; i < B_ * PROJ_;  i += stride) hbf[i] = (__bf16)0.0f;
}

// ----------------------------- phase 1: Zpre = x@W + bias ------------------
// wave w = tp*1024 + nt ; tp = timestep, nt = 16-col tile of 4U.
__global__ void __launch_bounds__(256) zpre_gemm_kernel(
        const __bf16* __restrict__ xbf, const __bf16* __restrict__ kpk,
        const float* __restrict__ bias, __bf16* __restrict__ zpre) {
    int lane = threadIdx.x & 31;
    int w    = blockIdx.x * 8 + (threadIdx.x >> 5);
    int tp   = w >> 10;       // 0..127
    int nt   = w & 1023;      // 0..1023
    const int KT = D_ / 32;   // 16

    const __bf16* abase = xbf + (size_t)tp * 32 * D_;
    const __bf16* bbase = kpk + (size_t)nt * KT * 512;

    f32x8 acc0 = {}, acc1 = {};
    for (int kt = 0; kt < KT; ++kt) {
        const __bf16* bf = bbase + (size_t)kt * 512;
        __builtin_prefetch(bf + 512, 0, 3);          // next frag, WGP-scope
        bf16x16 b  = load_b_frag(bf, lane);
        bf16x16 a0 = load_a_frag(abase, D_, 0,  kt * 32, lane);
        bf16x16 a1 = load_a_frag(abase, D_, 16, kt * 32, lane);
        acc0 = wmma_bf16(a0, b, acc0);
        acc1 = wmma_bf16(a1, b, acc1);
    }
    int n  = nt * 16 + (lane & 15);
    float bv = bias[n];
    int rb = (lane >> 4) << 3;
    __bf16* zr = zpre + (size_t)tp * B_ * G4_;
#pragma unroll
    for (int v = 0; v < 8; ++v) {
        zr[(size_t)(rb + v) * G4_ + n]      = (__bf16)(acc0[v] + bv);
        zr[(size_t)(16 + rb + v) * G4_ + n] = (__bf16)(acc1[v] + bv);
    }
}

// ----------------------------- per-step: gates ------------------------------
// 32 blocks x 256 threads; wave owns unit-tile ut (16 units, all 4 gates,
// both 16-row M tiles -> cell update is purely lane-local).
__global__ void __launch_bounds__(256) lstm_gates_kernel(
        const __bf16* __restrict__ zpre, const __bf16* __restrict__ rpk,
        const __bf16* __restrict__ hbf, float* __restrict__ cst,
        __bf16* __restrict__ hfull, int t) {
    int lane = threadIdx.x & 31;
    int ut   = blockIdx.x * 8 + (threadIdx.x >> 5);   // 0..255
    const int KT = PROJ_ / 32;                        // 16

    f32x8 acc[4][2] = {};
    for (int kt = 0; kt < KT; ++kt) {
        bf16x16 a0 = load_a_frag(hbf, PROJ_, 0,  kt * 32, lane);
        bf16x16 a1 = load_a_frag(hbf, PROJ_, 16, kt * 32, lane);
#pragma unroll
        for (int g = 0; g < 4; ++g) {
            const __bf16* bf = rpk + (((size_t)(g * 256 + ut)) * KT + kt) * 512;
            __builtin_prefetch(bf + 512, 0, 3);      // next frag, WGP-scope
            bf16x16 b = load_b_frag(bf, lane);
            acc[g][0] = wmma_bf16(a0, b, acc[g][0]);
            acc[g][1] = wmma_bf16(a1, b, acc[g][1]);
        }
    }

    int u = ut * 16 + (lane & 15);
    const __bf16* zr = zpre + (size_t)t * B_ * G4_;
#pragma unroll
    for (int mt = 0; mt < 2; ++mt) {
#pragma unroll
        for (int v = 0; v < 8; ++v) {
            int m = mt * 16 + ((lane >> 4) << 3) + v;
            size_t zb = (size_t)m * G4_ + u;
            float zi = acc[0][mt][v] + (float)zr[zb];
            float zf = acc[1][mt][v] + (float)zr[zb + UNITS_];
            float zc = acc[2][mt][v] + (float)zr[zb + 2 * UNITS_];
            float zo = acc[3][mt][v] + (float)zr[zb + 3 * UNITS_];
            size_t ci = (size_t)m * UNITS_ + u;
            float cn = hsig(zf) * cst[ci] + hsig(zi) * fast_tanh(zc);
            float h  = hsig(zo) * fast_tanh(cn);   // uses UNCLIPPED c
            cst[ci]   = clip3(cn);
            hfull[ci] = (__bf16)h;
        }
    }
}

// ----------------------------- per-step: projection -------------------------
// 32 blocks (one 16-col tile each) x 8 waves splitting K=4096; LDS reduce.
__global__ void __launch_bounds__(256) lstm_proj_kernel(
        const __bf16* __restrict__ hfull, const __bf16* __restrict__ ppk,
        float* __restrict__ out, __bf16* __restrict__ hbf, int t) {
    __shared__ float red[8 * 512];
    int lane = threadIdx.x & 31;
    int wid  = threadIdx.x >> 5;
    int nt   = blockIdx.x;                 // 0..31
    const int KT = UNITS_ / 32;            // 128, split 16 per wave

    f32x8 acc0 = {}, acc1 = {};
    for (int kk = 0; kk < KT / 8; ++kk) {
        int kt = wid * (KT / 8) + kk;
        const __bf16* bf = ppk + ((size_t)nt * KT + kt) * 512;
        __builtin_prefetch(bf + 512, 0, 3);
        bf16x16 b  = load_b_frag(bf, lane);
        bf16x16 a0 = load_a_frag(hfull, UNITS_, 0,  kt * 32, lane);
        bf16x16 a1 = load_a_frag(hfull, UNITS_, 16, kt * 32, lane);
        acc0 = wmma_bf16(a0, b, acc0);
        acc1 = wmma_bf16(a1, b, acc1);
    }
#pragma unroll
    for (int v = 0; v < 8; ++v) {
        red[wid * 512 + v * 32 + lane]       = acc0[v];
        red[wid * 512 + 256 + v * 32 + lane] = acc1[v];
    }
    __syncthreads();
    if (wid < 2) {
        int mt = wid;
#pragma unroll
        for (int v = 0; v < 8; ++v) {
            float s = 0.0f;
#pragma unroll
            for (int w = 0; w < 8; ++w)
                s += red[w * 512 + mt * 256 + v * 32 + lane];
            s = clip3(s);
            int m = mt * 16 + ((lane >> 4) << 3) + v;   // batch row
            int n = nt * 16 + (lane & 15);
            out[((size_t)m * S_ + t) * PROJ_ + n] = s;  // [B,S,PROJ]
            hbf[(size_t)m * PROJ_ + n] = (__bf16)s;     // next-step h
        }
    }
}

// ----------------------------- launch --------------------------------------
extern "C" void kernel_launch(void* const* d_in, const int* in_sizes, int n_in,
                              void* d_out, int out_size, void* d_ws, size_t ws_size,
                              hipStream_t stream) {
    const float* x    = (const float*)d_in[0];
    const float* Wk   = (const float*)d_in[1];
    const float* Wr   = (const float*)d_in[2];
    const float* bias = (const float*)d_in[3];
    const float* Wp   = (const float*)d_in[4];
    float* out = (float*)d_out;

    char* ws = (char*)d_ws;
    size_t off = 0;
    auto carve = [&](size_t bytes) -> void* {
        void* p = ws + off;
        off += (bytes + 255) & ~(size_t)255;
        return p;
    };
    __bf16* zpre  = (__bf16*)carve((size_t)S_ * B_ * G4_ * 2);     // 128 MB
    __bf16* kpk   = (__bf16*)carve((size_t)D_ * G4_ * 2);          // 16 MB
    __bf16* rpk   = (__bf16*)carve((size_t)PROJ_ * G4_ * 2);       // 16 MB
    __bf16* ppk   = (__bf16*)carve((size_t)UNITS_ * PROJ_ * 2);    // 4 MB
    __bf16* xbf   = (__bf16*)carve((size_t)S_ * B_ * D_ * 2);      // 4 MB
    float*  cst   = (float*)carve((size_t)B_ * UNITS_ * 4);        // 512 KB
    __bf16* hfull = (__bf16*)carve((size_t)B_ * UNITS_ * 2);       // 256 KB
    __bf16* hbf   = (__bf16*)carve((size_t)B_ * PROJ_ * 2);        // 32 KB

    // phase 0: prep (re-done every call: deterministic, graph-safe)
    xprep_kernel<<<256, 256, 0, stream>>>(x, xbf);
    pack_b_kernel<<<1024, 256, 0, stream>>>(Wk, kpk, D_, G4_);
    pack_b_kernel<<<1024, 256, 0, stream>>>(Wr, rpk, PROJ_, G4_);
    pack_b_kernel<<<256, 256, 0, stream>>>(Wp, ppk, UNITS_, PROJ_);
    init_state_kernel<<<256, 256, 0, stream>>>(cst, hbf);

    // phase 1: Zpre = x@W + bias   (M=4096, N=16384, K=512)
    zpre_gemm_kernel<<<16384, 256, 0, stream>>>(xbf, kpk, bias, zpre);

    // phase 2: recurrence (stream order provides the cross-step dependency)
    for (int t = 0; t < S_; ++t) {
        lstm_gates_kernel<<<32, 256, 0, stream>>>(zpre, rpk, hbf, cst, hfull, t);
        lstm_proj_kernel<<<32, 256, 0, stream>>>(hfull, ppk, out, hbf, t);
    }
}
